// Decoder_61057255080458
// MI455X (gfx1250) — compile-verified, matrix operand order checked by reading, not run on hardware
//
#include <hip/hip_runtime.h>
#include <hip/hip_bf16.h>

// ---------------------------------------------------------------------------
// Types for CDNA5 WMMA (gfx1250, wave32)
// ---------------------------------------------------------------------------
typedef __attribute__((ext_vector_type(16))) __bf16 v16bf;
typedef __attribute__((ext_vector_type(8)))  float  v8f;

union FragBF { v16bf v; uint4 q[2]; };

__device__ __forceinline__ unsigned short f2bf(float x) {
    unsigned int u = __float_as_uint(x);
    u += 0x7FFFu + ((u >> 16) & 1u);   // round-to-nearest-even
    return (unsigned short)(u >> 16);
}

__device__ __forceinline__ float blockReduceSum(float v, float* red) {
    int tid = threadIdx.x;
    red[tid] = v; __syncthreads();
    for (int s = 128; s > 0; s >>= 1) { if (tid < s) red[tid] += red[tid + s]; __syncthreads(); }
    float r = red[0]; __syncthreads(); return r;
}
__device__ __forceinline__ float blockReduceMax(float v, float* red) {
    int tid = threadIdx.x;
    red[tid] = v; __syncthreads();
    for (int s = 128; s > 0; s >>= 1) { if (tid < s) red[tid] = fmaxf(red[tid], red[tid + s]); __syncthreads(); }
    float r = red[0]; __syncthreads(); return r;
}

// Problem constants
#define H      512
#define E      512
#define T      512
#define BB     64
#define NLV    30000
#define OOV    64
#define VEXT   (NLV + OOV)          // 30064
#define X3H    1536                 // 3*H, also E+2H

// ---------------------------------------------------------------------------
// 1) embedding gather: embedded[b][e] = embedding[inputs[b]][e]
// ---------------------------------------------------------------------------
__global__ __launch_bounds__(256) void k_embed(const int* __restrict__ inputs,
                                               const float* __restrict__ emb,
                                               float* __restrict__ out) {
    int b = blockIdx.x;
    size_t row = (size_t)inputs[b] * E;
    for (int i = threadIdx.x; i < E; i += 256) out[b * E + i] = emb[row + i];
}

// ---------------------------------------------------------------------------
// 2) hW[b][j] = attn_b[j] + sum_k h0[b][k] * attn_W[j][k]   (first-half cols)
// ---------------------------------------------------------------------------
__global__ __launch_bounds__(256) void k_hW(const float* __restrict__ h0,
                                            const float* __restrict__ W,
                                            const float* __restrict__ bias,
                                            float* __restrict__ hW) {
    __shared__ float hs[H];
    int b = blockIdx.x;
    for (int i = threadIdx.x; i < H; i += 256) hs[i] = h0[b * H + i];
    __syncthreads();
    for (int j = threadIdx.x; j < H; j += 256) {
        const float* wr = W + (size_t)j * (2 * H);
        float acc = bias[j];
        for (int k = 0; k < H; ++k) acc += hs[k] * wr[k];
        hW[b * H + j] = acc;
    }
}

// ---------------------------------------------------------------------------
// 3) bf16 copy of W_e = attn_W[:, H:2H]   -> Wbf[j][k]
// ---------------------------------------------------------------------------
__global__ __launch_bounds__(256) void k_wbf(const float* __restrict__ W,
                                             unsigned short* __restrict__ Wbf) {
    int i = blockIdx.x * 256 + threadIdx.x;      // 512*512 elements
    int j = i >> 9, k = i & 511;
    Wbf[i] = f2bf(W[(size_t)j * (2 * H) + H + k]);
}

// ---------------------------------------------------------------------------
// 4) attention scores via WMMA bf16, fused relu + dot(v) epilogue
//    grid: (256 row-blocks, 3 attentions); block = 256 thr = 8 waves
//    each block: 128 rows (t*B+b) of enc staged bf16 in LDS, loop 32 j-tiles
// ---------------------------------------------------------------------------
__global__ __launch_bounds__(256) void k_attn_scores(
        const float* __restrict__ enc0, const float* __restrict__ enc1,
        const float* __restrict__ enc2,
        const unsigned short* __restrict__ Wbf,
        const float* __restrict__ hW, const float* __restrict__ v,
        float* __restrict__ scores) {
    __shared__ __align__(16) unsigned short Abuf[128 * 512];  // 128 KB
    __shared__ __align__(16) unsigned short Bbuf[16 * 512];   // 16 KB

    const float* enc = (blockIdx.y == 0) ? enc0 : ((blockIdx.y == 1) ? enc1 : enc2);
    const int rowBase = blockIdx.x * 128;

    // stage 128 rows x 512 f32 -> bf16 LDS (contiguous 256KB chunk of enc)
    const float4* src = (const float4*)(enc + (size_t)rowBase * H);
    for (int i = threadIdx.x; i < 128 * 512 / 4; i += 256) {
        float4 f = src[i];
        uint2 p;
        p.x = (unsigned)f2bf(f.x) | ((unsigned)f2bf(f.y) << 16);
        p.y = (unsigned)f2bf(f.z) | ((unsigned)f2bf(f.w) << 16);
        ((uint2*)Abuf)[i] = p;
    }

    const int lane = threadIdx.x & 31;
    const int wv   = threadIdx.x >> 5;
    const int mrow = lane & 15;      // A: M row  /  B,C: N column
    const int half = lane >> 4;
    const int kb   = half * 8;
    const int tileRow0 = rowBase + wv * 16;

    float rowsum[8];
#pragma unroll
    for (int r = 0; r < 8; ++r) rowsum[r] = 0.f;

    for (int jt = 0; jt < 32; ++jt) {
        const int jbase = jt * 16;
        __syncthreads();   // Abuf ready (iter0) / Bbuf consumers done (iter>0)
        for (int i = threadIdx.x; i < 16 * 512 / 2; i += 256)
            ((unsigned int*)Bbuf)[i] =
                ((const unsigned int*)(Wbf + (size_t)jbase * 512))[i];
        __syncthreads();

        v8f acc = {0.f, 0.f, 0.f, 0.f, 0.f, 0.f, 0.f, 0.f};
#pragma unroll
        for (int ks = 0; ks < 16; ++ks) {
            const int k0 = ks * 32 + kb;
            FragBF a, bf;
            const uint4* ap = (const uint4*)&Abuf[(wv * 16 + mrow) * 512 + k0];
            a.q[0] = ap[0]; a.q[1] = ap[2];          // K [kb..kb+8), [kb+16..kb+24)
            const uint4* bp = (const uint4*)&Bbuf[mrow * 512 + k0];
            bf.q[0] = bp[0]; bf.q[1] = bp[2];
            acc = __builtin_amdgcn_wmma_f32_16x16x32_bf16(
                false, a.v, false, bf.v, (short)0, acc, false, false);
        }
        // epilogue: score += v[j] * relu(acc + hW[b][j])
        const int j = jbase + mrow;
        const float vj = v[j];
#pragma unroll
        for (int r = 0; r < 8; ++r) {
            int row = tileRow0 + r + 8 * half;
            int b_ = row & 63;
            float e = acc[r] + hW[b_ * H + j];
            rowsum[r] += vj * fmaxf(e, 0.f);
        }
    }
    // reduce over N (16 lanes in each half), write score[b][t]
#pragma unroll
    for (int r = 0; r < 8; ++r) {
        float s = rowsum[r];
        s += __shfl_xor(s, 1); s += __shfl_xor(s, 2);
        s += __shfl_xor(s, 4); s += __shfl_xor(s, 8);
        if (mrow == 0) {
            int row = tileRow0 + r + 8 * half;
            int t = row >> 6, b_ = row & 63;
            scores[(size_t)blockIdx.y * (BB * T) + b_ * T + t] = s;
        }
    }
}

// ---------------------------------------------------------------------------
// 5) softmax over T (in place); grid 3*64 rows
// ---------------------------------------------------------------------------
__global__ __launch_bounds__(256) void k_softmaxT(float* __restrict__ scores) {
    __shared__ float red[256];
    float* s = scores + (size_t)blockIdx.x * T;
    float v0 = s[threadIdx.x], v1 = s[threadIdx.x + 256];
    float mx = blockReduceMax(fmaxf(v0, v1), red);
    float e0 = expf(v0 - mx), e1 = expf(v1 - mx);
    float sum = blockReduceSum(e0 + e1, red);
    s[threadIdx.x] = e0 / sum; s[threadIdx.x + 256] = e1 / sum;
}

// ---------------------------------------------------------------------------
// 6) ctx[a][b][h] = sum_t w[a][b][t] * enc[t][b][h]; grid (64, 3)
// ---------------------------------------------------------------------------
__global__ __launch_bounds__(256) void k_ctx(
        const float* __restrict__ enc0, const float* __restrict__ enc1,
        const float* __restrict__ enc2, const float* __restrict__ w,
        float* __restrict__ ctx) {
    __shared__ float wsm[T];
    int b = blockIdx.x, a = blockIdx.y;
    const float* enc = (a == 0) ? enc0 : ((a == 1) ? enc1 : enc2);
    for (int i = threadIdx.x; i < T; i += 256)
        wsm[i] = w[(size_t)a * (BB * T) + b * T + i];
    __syncthreads();
    int h0 = threadIdx.x, h1 = threadIdx.x + 256;
    float a0 = 0.f, a1 = 0.f;
    for (int t = 0; t < T; ++t) {
        const float* er = enc + (size_t)t * (BB * H) + b * H;
        float wt = wsm[t];
        a0 += wt * er[h0]; a1 += wt * er[h1];
    }
    ctx[(size_t)a * (BB * H) + b * H + h0] = a0;
    ctx[(size_t)a * (BB * H) + b * H + h1] = a1;
}

// ---------------------------------------------------------------------------
// 7) small fused kernel per batch row: poss, context, p_gen, GRU, h_new
// ---------------------------------------------------------------------------
__global__ __launch_bounds__(256) void k_small(
        const float* __restrict__ h0g, const float* __restrict__ embedded,
        const float* __restrict__ ctx,
        const float* __restrict__ pickW, const float* __restrict__ pickb,
        const float* __restrict__ pgenW, const float* __restrict__ pgenb,
        const float* __restrict__ Wih, const float* __restrict__ Whh,
        const float* __restrict__ bih, const float* __restrict__ bhh,
        float* __restrict__ contextOut, float* __restrict__ pgenOut,
        unsigned short* __restrict__ xbf, float* __restrict__ hnewOut) {
    __shared__ float hs[H], es[E], cs[2 * H], xs[X3H], red[256];
    int b = blockIdx.x, tid = threadIdx.x;
    for (int i = tid; i < H; i += 256) { hs[i] = h0g[b * H + i]; es[i] = embedded[b * E + i]; }
    __syncthreads();

    // poss = sigmoid(softmax(h0) . pickW + pickb)
    float lm = -1e30f;
    for (int i = tid; i < H; i += 256) lm = fmaxf(lm, hs[i]);
    float hm = blockReduceMax(lm, red);
    float le = 0.f, ld = 0.f;
    for (int i = tid; i < H; i += 256) {
        float ev = expf(hs[i] - hm);
        le += ev; ld += ev * pickW[i];
    }
    float se = blockReduceSum(le, red);
    float sd = blockReduceSum(ld, red);
    float poss = 1.f / (1.f + expf(-(sd / se + pickb[0])));

    // context = [gat_ctx, code_ctx + poss*tree_ctx]
    const float* codec = ctx + 0 * (BB * H) + b * H;
    const float* gatc  = ctx + 1 * (BB * H) + b * H;
    const float* treec = ctx + 2 * (BB * H) + b * H;
    for (int i = tid; i < H; i += 256) {
        cs[i]     = gatc[i];
        cs[H + i] = codec[i] + poss * treec[i];
    }
    __syncthreads();

    // p_gen = sigmoid([context, h0, embedded] . pgenW + pgenb)
    float pga = 0.f;
    for (int k = tid; k < 2 * H + H + E; k += 256) {
        float xv = (k < 2 * H) ? cs[k] : ((k < 3 * H) ? hs[k - 2 * H] : es[k - 3 * H]);
        pga += xv * pgenW[k];
    }
    float ps = blockReduceSum(pga, red);
    float pg = 1.f / (1.f + expf(-(ps + pgenb[0])));
    if (tid == 0) pgenOut[b] = pg;

    // x = [embedded, context]
    for (int i = tid; i < E; i += 256) xs[i] = es[i];
    for (int i = tid; i < 2 * H; i += 256) {
        xs[E + i] = cs[i];
        contextOut[b * (2 * H) + i] = cs[i];
        xbf[(size_t)b * X3H + H + i] = f2bf(cs[i]);   // [h_new|context] row, ctx part
    }
    __syncthreads();

    // single-step GRU (gate order r,z,n)
    for (int j = tid; j < H; j += 256) {
        float aR = bih[j], aZ = bih[j + H], aN = bih[j + 2 * H];
        const float* wr = Wih + (size_t)j * X3H;
        const float* wz = Wih + (size_t)(j + H) * X3H;
        const float* wn = Wih + (size_t)(j + 2 * H) * X3H;
        for (int k = 0; k < X3H; ++k) {
            float xv = xs[k];
            aR += xv * wr[k]; aZ += xv * wz[k]; aN += xv * wn[k];
        }
        float hR = bhh[j], hZ = bhh[j + H], hN = bhh[j + 2 * H];
        const float* vr = Whh + (size_t)j * H;
        const float* vz = Whh + (size_t)(j + H) * H;
        const float* vn = Whh + (size_t)(j + 2 * H) * H;
        for (int k = 0; k < H; ++k) {
            float hv = hs[k];
            hR += hv * vr[k]; hZ += hv * vz[k]; hN += hv * vn[k];
        }
        float r = 1.f / (1.f + expf(-(aR + hR)));
        float z = 1.f / (1.f + expf(-(aZ + hZ)));
        float n = tanhf(aN + r * hN);
        float hn = (1.f - z) * n + z * hs[j];
        hnewOut[b * H + j] = hn;
        xbf[(size_t)b * X3H + j] = f2bf(hn);
    }
}

// ---------------------------------------------------------------------------
// 8) logits = [h_new|context] @ out_W^T + out_b, via WMMA bf16
//    M=64 (4 tiles per wave), one 16-col N-tile per wave, K=1536 (48 steps)
// ---------------------------------------------------------------------------
__global__ __launch_bounds__(256) void k_logits(
        const unsigned short* __restrict__ xbf, const float* __restrict__ outW,
        const float* __restrict__ outb, float* __restrict__ logits) {
    const int lane = threadIdx.x & 31, wv = threadIdx.x >> 5;
    const int nt = blockIdx.x * 8 + wv;
    if (nt >= NLV / 16) return;                 // wave-uniform
    const int nbase = nt * 16;
    const int nr = lane & 15, half = lane >> 4, kb = half * 8;

    v8f acc[4];
#pragma unroll
    for (int mt = 0; mt < 4; ++mt) acc[mt] = (v8f){0.f,0.f,0.f,0.f,0.f,0.f,0.f,0.f};

    const float* wrow = outW + (size_t)(nbase + nr) * X3H;
    for (int ks = 0; ks < 48; ++ks) {
        const int k0 = ks * 32 + kb;
        FragBF bf;
        {
            float4 f0 = *(const float4*)(wrow + k0);
            float4 f1 = *(const float4*)(wrow + k0 + 4);
            float4 f2 = *(const float4*)(wrow + k0 + 16);
            float4 f3 = *(const float4*)(wrow + k0 + 20);
            bf.q[0].x = (unsigned)f2bf(f0.x) | ((unsigned)f2bf(f0.y) << 16);
            bf.q[0].y = (unsigned)f2bf(f0.z) | ((unsigned)f2bf(f0.w) << 16);
            bf.q[0].z = (unsigned)f2bf(f1.x) | ((unsigned)f2bf(f1.y) << 16);
            bf.q[0].w = (unsigned)f2bf(f1.z) | ((unsigned)f2bf(f1.w) << 16);
            bf.q[1].x = (unsigned)f2bf(f2.x) | ((unsigned)f2bf(f2.y) << 16);
            bf.q[1].y = (unsigned)f2bf(f2.z) | ((unsigned)f2bf(f2.w) << 16);
            bf.q[1].z = (unsigned)f2bf(f3.x) | ((unsigned)f2bf(f3.y) << 16);
            bf.q[1].w = (unsigned)f2bf(f3.z) | ((unsigned)f2bf(f3.w) << 16);
        }
#pragma unroll
        for (int mt = 0; mt < 4; ++mt) {
            FragBF a;
            const uint4* ap = (const uint4*)(xbf + (size_t)(mt * 16 + nr) * X3H + k0);
            a.q[0] = ap[0]; a.q[1] = ap[2];
            acc[mt] = __builtin_amdgcn_wmma_f32_16x16x32_bf16(
                false, a.v, false, bf.v, (short)0, acc[mt], false, false);
        }
    }
    const float bn = outb[nbase + nr];
#pragma unroll
    for (int mt = 0; mt < 4; ++mt)
#pragma unroll
        for (int r = 0; r < 8; ++r) {
            int m = mt * 16 + r + 8 * half;
            logits[(size_t)m * NLV + nbase + nr] = acc[mt][r] + bn;
        }
}

// ---------------------------------------------------------------------------
// 9) per-row max & sum(exp) over 30000
// ---------------------------------------------------------------------------
__global__ __launch_bounds__(256) void k_rowstats(const float* __restrict__ logits,
                                                  float* __restrict__ rmax,
                                                  float* __restrict__ rsum) {
    __shared__ float red[256];
    int m = blockIdx.x;
    const float* row = logits + (size_t)m * NLV;
    float mx = -1e30f;
    for (int n = threadIdx.x; n < NLV; n += 256) mx = fmaxf(mx, row[n]);
    mx = blockReduceMax(mx, red);
    float s = 0.f;
    for (int n = threadIdx.x; n < NLV; n += 256) s += expf(row[n] - mx);
    s = blockReduceSum(s, red);
    if (threadIdx.x == 0) { rmax[m] = mx; rsum[m] = s; }
}

// ---------------------------------------------------------------------------
// 10) final: p_gen mix + deterministic scatter-add + log
// ---------------------------------------------------------------------------
__global__ __launch_bounds__(256) void k_final(
        const float* __restrict__ logits, const float* __restrict__ rmax,
        const float* __restrict__ rsum, const float* __restrict__ pgen,
        const float* __restrict__ code_w, const int* __restrict__ ext_idx,
        float* __restrict__ out) {
    __shared__ int   idxs[T];
    __shared__ float avals[T];
    int m = blockIdx.x;
    float pg = pgen[m], mx = rmax[m], sm = rsum[m];
    for (int t = threadIdx.x; t < T; t += 256) {
        idxs[t]  = ext_idx[m * T + t];
        avals[t] = (1.f - pg) * code_w[m * T + t];
    }
    float* orow = out + (size_t)m * VEXT;
    const float* lrow = logits + (size_t)m * NLV;
    for (int n = threadIdx.x; n < VEXT; n += 256)
        orow[n] = (n < NLV) ? pg * expf(lrow[n] - mx) / sm : 0.f;
    __threadfence(); __syncthreads();
    // each thread owns disjoint column range -> deterministic scatter
    int lo = threadIdx.x * 118, hi = lo + 118;
    if (hi > VEXT) hi = VEXT;
    for (int t = 0; t < T; ++t) {
        int id = idxs[t];
        if (id >= lo && id < hi) orow[id] += avals[t];
    }
    __threadfence(); __syncthreads();
    for (int n = threadIdx.x; n < VEXT; n += 256)
        orow[n] = logf(orow[n] + 1e-12f);
}

// ---------------------------------------------------------------------------
// launch
// ---------------------------------------------------------------------------
extern "C" void kernel_launch(void* const* d_in, const int* in_sizes, int n_in,
                              void* d_out, int out_size, void* d_ws, size_t ws_size,
                              hipStream_t stream) {
    const int*   inputs   = (const int*)  d_in[0];
    const float* last_h   = (const float*)d_in[1];   // [1,B,H]
    const float* code_out = (const float*)d_in[2];
    const float* gat_out  = (const float*)d_in[3];
    const float* tree_out = (const float*)d_in[4];
    const int*   ext_idx  = (const int*)  d_in[5];
    const float* emb      = (const float*)d_in[7];
    const float* attn_W   = (const float*)d_in[8];
    const float* attn_b   = (const float*)d_in[9];
    const float* attn_v   = (const float*)d_in[10];
    const float* pick_W   = (const float*)d_in[11];
    const float* pick_b   = (const float*)d_in[12];
    const float* pgen_W   = (const float*)d_in[13];
    const float* pgen_b   = (const float*)d_in[14];
    const float* gru_Wih  = (const float*)d_in[15];
    const float* gru_Whh  = (const float*)d_in[16];
    const float* gru_bih  = (const float*)d_in[17];
    const float* gru_bhh  = (const float*)d_in[18];
    const float* out_W    = (const float*)d_in[19];
    const float* out_b    = (const float*)d_in[20];

    char* ws = (char*)d_ws;
    float*          embedded = (float*)(ws + 0);                       // 128 KB
    float*          hW       = (float*)(ws + (131072));                // 128 KB
    unsigned short* Wbf      = (unsigned short*)(ws + (262144));       // 512 KB
    float*          scores   = (float*)(ws + (786432));                // 384 KB
    float*          ctx      = (float*)(ws + (1179648));               // 384 KB
    float*          context  = (float*)(ws + (1572864));               // 256 KB
    float*          pgenv    = (float*)(ws + (1835008));               // 1 KB
    unsigned short* xbf      = (unsigned short*)(ws + (1836032));      // 192 KB
    float*          logits   = (float*)(ws + (2097152));               // 7.68 MB
    float*          rmax     = (float*)(ws + (2097152 + 7680000));
    float*          rsum     = (float*)(ws + (2097152 + 7680000 + 256));

    float* final_out = (float*)d_out;                 // [64,30064]
    float* hnew_out  = final_out + (size_t)BB * VEXT; // [1,64,512]

    k_embed<<<BB, 256, 0, stream>>>(inputs, emb, embedded);
    k_hW<<<BB, 256, 0, stream>>>(last_h, attn_W, attn_b, hW);
    k_wbf<<<(H * H) / 256, 256, 0, stream>>>(attn_W, Wbf);

    k_attn_scores<<<dim3((T * BB) / 128, 3), 256, 0, stream>>>(
        code_out, gat_out, tree_out, Wbf, hW, attn_v, scores);

    k_softmaxT<<<3 * BB, 256, 0, stream>>>(scores);

    k_ctx<<<dim3(BB, 3), 256, 0, stream>>>(code_out, gat_out, tree_out, scores, ctx);

    k_small<<<BB, 256, 0, stream>>>(last_h, embedded, ctx, pick_W, pick_b,
                                    pgen_W, pgen_b, gru_Wih, gru_Whh, gru_bih,
                                    gru_bhh, context, pgenv, xbf, hnew_out);

    k_logits<<<(NLV / 16 + 7) / 8, 256, 0, stream>>>(xbf, out_W, out_b, logits);

    k_rowstats<<<BB, 256, 0, stream>>>(logits, rmax, rsum);

    k_final<<<BB, 256, 0, stream>>>(logits, rmax, rsum, pgenv,
                                    scores /* attn 0 = code weights */,
                                    ext_idx, final_out);
}
